// GACFV4_layer_75196287419132
// MI455X (gfx1250) — matrix operand order repaired and checked by reading, not compiled, not run on hardware
//
#include <hip/hip_runtime.h>

// ---------------------------------------------------------------------------
// GACF layer for MI455X (gfx1250, wave32, WMMA).
// Dense math on v_wmma_f32_16x16x32_bf16 (fp32 accumulate).
// K / V^T are materialized in bf16 so the attention inner loop does raw
// b128 loads (no per-iteration cvt). Flash-style online softmax keeps the
// 3072x3072 score matrices in registers.
// ---------------------------------------------------------------------------

#define U_NUM   3072
#define I_NUM   3072
#define N_NODES 6144
#define EMBED   128
#define NNZ     393216
#define HEADS   8
#define DH      16

typedef __attribute__((ext_vector_type(16))) __bf16 v16bf;
typedef __attribute__((ext_vector_type(8)))  __bf16 v8bf;
typedef __attribute__((ext_vector_type(8)))  float  v8f;

__device__ __forceinline__ __bf16 tobf(float x) { return (__bf16)x; }

__device__ __forceinline__ v16bf cat8(v8bf lo, v8bf hi) {
    return __builtin_shufflevector(lo, hi, 0, 1, 2, 3, 4, 5, 6, 7,
                                           8, 9, 10, 11, 12, 13, 14, 15);
}

// --------------------------------------------------------------------------
// features = concat(uEmbd, iEmbd); agg1 starts as a copy (fuses "+ features")
// --------------------------------------------------------------------------
__global__ void k_concat(const float* __restrict__ u, const float* __restrict__ it,
                         float* __restrict__ ffeat, float* __restrict__ agg) {
    int idx = blockIdx.x * 256 + threadIdx.x;      // 0 .. N_NODES*EMBED-1
    int n = idx >> 7;
    float v = (n < U_NUM) ? u[idx] : it[idx - U_NUM * EMBED];
    ffeat[idx] = v;
    agg[idx]   = v;
}

__global__ void k_zero(float* __restrict__ p, int n) {
    int i = blockIdx.x * 256 + threadIdx.x;
    if (i < n) p[i] = 0.0f;
}

// --------------------------------------------------------------------------
// SPMM: agg[row] += val * x[col]   (32 lanes per nnz, float4 gather)
// --------------------------------------------------------------------------
__global__ void k_spmm(const float* __restrict__ vals, const int* __restrict__ rows,
                       const int* __restrict__ cols, const float* __restrict__ x,
                       float* __restrict__ agg) {
    int t  = blockIdx.x * 256 + threadIdx.x;       // NNZ * 32 threads
    int nz = t >> 5;
    int ln = t & 31;
    int r = rows[nz], c = cols[nz];
    float v = vals[nz];
    float4 xv = ((const float4*)(x + (size_t)c * EMBED))[ln];
    float* dst = agg + (size_t)r * EMBED + ln * 4;
    atomicAdd(dst + 0, v * xv.x);
    atomicAdd(dst + 1, v * xv.y);
    atomicAdd(dst + 2, v * xv.z);
    atomicAdd(dst + 3, v * xv.w);
}

// --------------------------------------------------------------------------
// out[M x 128] = epilogue( X[M x 128] @ W[128 out][128 in]^T + b )
// mode: 0 = none(fp32), 1 = relu(fp32), 2 = square(fp32), 3 = store bf16
// One wave -> one 16x16 D tile; K=128 as 4 x wmma_f32_16x16x32_bf16.
// --------------------------------------------------------------------------
__global__ void k_gemm128(const float* __restrict__ X, const float* __restrict__ W,
                          const float* __restrict__ bias, float* __restrict__ out,
                          __bf16* __restrict__ outbf, int mode) {
    int wave = blockIdx.x * 4 + (threadIdx.x >> 5);
    int lane = threadIdx.x & 31;
    int lm = lane & 15, half = lane >> 4;
    int tm = wave >> 3;          // 384 row tiles
    int tn = wave & 7;           // 8 col tiles

    const float* xrow = X + (size_t)(tm * 16 + lm) * EMBED;   // A: lane = M
    const float* wrow = W + (size_t)(tn * 16 + lm) * EMBED;   // B: lane = N

    v8f acc = {};
    #pragma unroll
    for (int kk = 0; kk < 128; kk += 32) {
        v16bf a, b;
        // A 16x32 bf16: half 0 -> K 0..15, half 1 -> K 16..31 (contiguous)
        #pragma unroll
        for (int e = 0; e < 8; ++e) {
            a[e]     = tobf(xrow[kk + half * 16 + e]);
            a[8 + e] = tobf(xrow[kk + half * 16 + 8 + e]);
        }
        // B 32x16 bf16: half 0 -> K 0..15, half 1 -> K 16..31 (contiguous)
        #pragma unroll
        for (int e = 0; e < 16; ++e)
            b[e] = tobf(wrow[kk + half * 16 + e]);
        acc = __builtin_amdgcn_wmma_f32_16x16x32_bf16(false, a, false, b,
                                                      (short)0, acc, false, false);
    }

    int col = tn * 16 + lm;
    float bv = bias[col];
    #pragma unroll
    for (int r = 0; r < 8; ++r) {                  // C layout: M = r + 8*half
        int row = tm * 16 + half * 8 + r;
        float v = acc[r] + bv;
        if (mode == 3) {
            outbf[(size_t)row * EMBED + col] = tobf(v);
        } else {
            if (mode == 1)      v = v > 0.0f ? v : 0.0f;
            else if (mode == 2) v = v * v;
            out[(size_t)row * EMBED + col] = v;
        }
    }
}

// --------------------------------------------------------------------------
// Vt[d_global][node] = V[node][d_global]  (bf16; B-tiles for P@V want K contig)
// --------------------------------------------------------------------------
__global__ void k_transpose(const __bf16* __restrict__ V, __bf16* __restrict__ Vt) {
    int idx = blockIdx.x * 256 + threadIdx.x;      // N_NODES*EMBED threads
    int d = idx / N_NODES;
    int n = idx - d * N_NODES;
    Vt[idx] = V[(size_t)n * EMBED + d];
}

// --------------------------------------------------------------------------
// Flash attention: one wave per (direction, head, 16-query tile).
// Per iteration: 32 keys -> 2 QK^T wmmas, online softmax, P staged via LDS
// (same-wave DS ops are in-order on CDNA5), 1 P@V wmma with full K=32.
// K / V^T are bf16 in memory: B-tiles are raw b128 loads, zero cvt.
// --------------------------------------------------------------------------
__global__ void k_attn(const float* __restrict__ Q, const __bf16* __restrict__ Kb,
                       const __bf16* __restrict__ Vt, float* __restrict__ attn) {
    __shared__ float pT[4][16][32];                // per-wave 16x32 P tile
    int widx = threadIdx.x >> 5;
    int wave = blockIdx.x * 4 + widx;              // 3072 waves total
    int lane = threadIdx.x & 31;
    int lm = lane & 15, half = lane >> 4;

    int dir  = wave / 1536;                        // 0: users->items, 1: items->users
    int rr   = wave - dir * 1536;
    int head = rr / 192;
    int qt   = rr - head * 192;
    int q0   = dir * U_NUM + qt * 16;              // query node base
    int kv0  = (1 - dir) * U_NUM;                  // key/value node base

    // Q A-tile (16 x dh, K padded 16 -> 32 with zeros), loop-invariant
    v16bf qa;
    {
        const float* qrow = Q + (size_t)(q0 + lm) * EMBED + head * DH;
        #pragma unroll
        for (int e = 0; e < 8; ++e) {
            qa[e]     = half ? (__bf16)0.0f : tobf(qrow[e]);      // K = e (half 0); half 1 pad
            qa[8 + e] = half ? (__bf16)0.0f : tobf(qrow[8 + e]);  // K = 8 + e (half 0); half 1 pad
        }
    }

    float mrow[8], lrow[8];
    #pragma unroll
    for (int r = 0; r < 8; ++r) { mrow[r] = -1e30f; lrow[r] = 0.0f; }
    v8f acc = {};
    const float scale = 0.25f;                     // 1/sqrt(dh)

    for (int kk = 0; kk < U_NUM; kk += 32) {
        // K B-tiles: 16 keys each; K-dim(= d) 0..15 real, 16..31 zero pad
        v16bf b1 = {}, b2 = {};
        if (half == 0) {
            const v8bf* k1 = (const v8bf*)(Kb + (size_t)(kv0 + kk + lm) * EMBED + head * DH);
            const v8bf* k2 = (const v8bf*)((const __bf16*)k1 + 16 * EMBED);
            b1 = cat8(k1[0], k1[1]);
            b2 = cat8(k2[0], k2[1]);
            if (kk + 32 < U_NUM) {                 // prefetch next key tile (L2-resident)
                __builtin_prefetch((const __bf16*)k1 + 32 * EMBED, 0, 3);
                __builtin_prefetch((const __bf16*)k2 + 32 * EMBED, 0, 3);
            }
        }
        v8f z = {};
        v8f S1 = __builtin_amdgcn_wmma_f32_16x16x32_bf16(false, qa, false, b1,
                                                         (short)0, z, false, false);
        v8f S2 = __builtin_amdgcn_wmma_f32_16x16x32_bf16(false, qa, false, b2,
                                                         (short)0, z, false, false);

        #pragma unroll
        for (int r = 0; r < 8; ++r) {              // row m = r + 8*half
            float f1 = S1[r] * scale, f2 = S2[r] * scale;
            float rm = fmaxf(f1, f2);
            #pragma unroll
            for (int m = 1; m < 16; m <<= 1) rm = fmaxf(rm, __shfl_xor(rm, m, 32));
            float mnew = fmaxf(mrow[r], rm);
            float corr = __expf(mrow[r] - mnew);
            mrow[r] = mnew;
            lrow[r] *= corr;
            acc[r]  *= corr;
            float p1 = __expf(f1 - mnew);
            float p2 = __expf(f2 - mnew);
            float rs = p1 + p2;
            #pragma unroll
            for (int m = 1; m < 16; m <<= 1) rs += __shfl_xor(rs, m, 32);
            lrow[r] += rs;
            pT[widx][half * 8 + r][lm]      = p1;  // col = key-in-tile
            pT[widx][half * 8 + r][16 + lm] = p2;
        }
        asm volatile("" ::: "memory");             // keep DS store->load order in IR

        // Re-layout P (C layout -> A layout, fp32 -> bf16); V B-tile raw load
        v16bf pa;
        #pragma unroll
        for (int e = 0; e < 8; ++e) {
            pa[e]     = tobf(pT[widx][lm][half * 16 + e]);
            pa[8 + e] = tobf(pT[widx][lm][half * 16 + 8 + e]);
        }
        const v8bf* vp = (const v8bf*)(Vt + (size_t)(head * DH + lm) * N_NODES
                                          + kv0 + kk + half * 16);
        v16bf bv = cat8(vp[0], vp[1]);

        acc = __builtin_amdgcn_wmma_f32_16x16x32_bf16(false, pa, false, bv,
                                                      (short)0, acc, false, false);
        asm volatile("" ::: "memory");
    }

    int col = head * DH + lm;
    #pragma unroll
    for (int r = 0; r < 8; ++r) {
        int row = q0 + half * 8 + r;
        attn[(size_t)row * EMBED + col] = acc[r] / lrow[r];
    }
}

// --------------------------------------------------------------------------
// out = concat([features, feature1 + feature2], axis=1)   (row stride 256)
// --------------------------------------------------------------------------
__global__ void k_final(const float* __restrict__ ffeat, const float* __restrict__ f1,
                        const float* __restrict__ f2, float* __restrict__ out) {
    int idx = blockIdx.x * 256 + threadIdx.x;      // N_NODES*EMBED
    int n = idx >> 7, c = idx & 127;
    out[(size_t)n * 256 + c]       = ffeat[idx];
    out[(size_t)n * 256 + 128 + c] = f1[idx] + f2[idx];
}

// --------------------------------------------------------------------------
extern "C" void kernel_launch(void* const* d_in, const int* in_sizes, int n_in,
                              void* d_out, int out_size, void* d_ws, size_t ws_size,
                              hipStream_t stream) {
    (void)in_sizes; (void)n_in; (void)out_size; (void)ws_size;
    const float* uE  = (const float*)d_in[2];
    const float* iE  = (const float*)d_in[3];
    const float* lv  = (const float*)d_in[4];
    const int*   lr  = (const int*)d_in[5];
    const int*   lc  = (const int*)d_in[6];
    const float* a1w = (const float*)d_in[7];
    const float* a1b = (const float*)d_in[8];
    const float* a2w = (const float*)d_in[9];
    const float* a2b = (const float*)d_in[10];
    const float* ipw = (const float*)d_in[11];
    const float* ipb = (const float*)d_in[12];
    const float* opw = (const float*)d_in[13];
    const float* opb = (const float*)d_in[14];
    float* out = (float*)d_out;
    float* ws  = (float*)d_ws;

    const size_t F = (size_t)N_NODES * EMBED;      // 786432 floats per plane
    float*  ffeat = ws + 0 * F;
    float*  agg   = ws + 1 * F;
    float*  feat1 = ws + 2 * F;
    float*  Qb    = ws + 3 * F;
    __bf16* Kbf   = (__bf16*)(ws + 4 * F);         // bf16: uses half a plane
    __bf16* Vbf   = (__bf16*)(ws + 5 * F);
    __bf16* Vtb   = (__bf16*)(ws + 6 * F);
    float*  attnb = ws + 7 * F;
    float*  inter = ws + 8 * F;
    float*  feat2 = ws + 9 * F;

    const int elemBlocks = (int)(F / 256);         // 3072
    const int spmmBlocks = NNZ * 32 / 256;         // 49152
    const int gemmBlocks = 768;                    // 3072 waves = 384x8 tiles

    // features, agg1 <- features
    k_concat<<<elemBlocks, 256, 0, stream>>>(uE, iE, ffeat, agg);
    // agg1 += L @ features
    k_spmm<<<spmmBlocks, 256, 0, stream>>>(lv, lr, lc, ffeat, agg);
    // feature1 = relu(agg1 @ a1.T + b1)
    k_gemm128<<<gemmBlocks, 128, 0, stream>>>(agg, a1w, a1b, feat1, nullptr, 1);
    // Q fp32; K, V bf16 (slices of in_proj)
    k_gemm128<<<gemmBlocks, 128, 0, stream>>>(ffeat, ipw,                 ipb,       Qb, nullptr, 0);
    k_gemm128<<<gemmBlocks, 128, 0, stream>>>(ffeat, ipw + 128 * 128,     ipb + 128, nullptr, Kbf, 3);
    k_gemm128<<<gemmBlocks, 128, 0, stream>>>(ffeat, ipw + 2 * 128 * 128, ipb + 256, nullptr, Vbf, 3);
    k_transpose<<<elemBlocks, 256, 0, stream>>>(Vbf, Vtb);
    // both cross-attentions (dir encoded in wave id)
    k_attn<<<768, 128, 0, stream>>>(Qb, Kbf, Vtb, attnb);
    // inter = (attn @ Wo.T + bo)^2
    k_gemm128<<<gemmBlocks, 128, 0, stream>>>(attnb, opw, opb, inter, nullptr, 2);
    // agg2 = L @ inter
    k_zero<<<elemBlocks, 256, 0, stream>>>(agg, (int)F);
    k_spmm<<<spmmBlocks, 256, 0, stream>>>(lv, lr, lc, inter, agg);
    // feature2 = relu(agg2 @ a2.T + b2)
    k_gemm128<<<gemmBlocks, 128, 0, stream>>>(agg, a2w, a2b, feat2, nullptr, 1);
    // out = [features | feature1 + feature2]
    k_final<<<elemBlocks, 256, 0, stream>>>(ffeat, feat1, feat2, out);
}
